// GCN_79267916415512
// MI455X (gfx1250) — matrix operand order repaired and checked by reading, not compiled, not run on hardware
//
#include <hip/hip_runtime.h>
#include <hip/hip_bf16.h>

typedef __attribute__((ext_vector_type(16))) __bf16 v16bf;
typedef __attribute__((ext_vector_type(8)))  float  v8f;

#define K_DIM  128
#define KSTEPS 4   // 128 / 32

// ---------------- degree / dinv ----------------
__global__ void init_deg_kernel(float* deg, int n) {
    int i = blockIdx.x * blockDim.x + threadIdx.x;
    if (i < n) deg[i] = 1.0f;                       // self-loop
}

__global__ void deg_edges_kernel(const int* __restrict__ dst, float* deg, int E) {
    int e = blockIdx.x * blockDim.x + threadIdx.x;
    if (e < E) atomicAdd(&deg[dst[e]], 1.0f);
}

__global__ void dinv_kernel(float* deg, int n) {
    int i = blockIdx.x * blockDim.x + threadIdx.x;
    if (i < n) deg[i] = rsqrtf(deg[i]);             // deg >= 1 always
}

// ---------------- WMMA GEMM: out[M,NOUT] = actv(A[M,128]) @ W[128,NOUT] ----------------
// actv = identity, or relu(x + bias[k]) when RELU (fused layer-1 epilogue into layer-2 A load)
template<int NTILES, int NOUT, bool RELU>
__global__ __launch_bounds__(128) void gemm_wmma_kernel(
    const float* __restrict__ A, const float* __restrict__ W,
    const float* __restrict__ bias, float* __restrict__ out, int mtiles)
{
    // B pre-swizzled in LDS: fragment for (kstep, ntile, lane) is 16 contiguous bf16 (32B)
    __shared__ alignas(32) __bf16 Blds[KSTEPS * NTILES * 32 * 16];
    __shared__ float biasS[K_DIM];

    for (int idx = threadIdx.x; idx < KSTEPS * NTILES * 32 * 16; idx += blockDim.x) {
        int i    = idx & 15;            // fragment element -> K within 16-row group
        int lane = (idx >> 4) & 31;
        int t    = (idx >> 9) % NTILES;
        int ks   = idx / (NTILES * 512);
        int k = ks * 32 + (lane >> 4) * 8 + (i < 8 ? i : i + 8);     // lanes 0-15: K {0..7,16..23}, lanes 16-31: K {8..15,24..31}
        int n = t * 16 + (lane & 15);
        float v = (n < NOUT) ? W[k * NOUT + n] : 0.0f;   // zero-pad cols (NOUT=40 -> 48)
        Blds[idx] = (__bf16)v;
    }
    if (RELU) {
        for (int k = threadIdx.x; k < K_DIM; k += blockDim.x) biasS[k] = bias[k];
    }
    __syncthreads();

    int tile = blockIdx.x * 4 + (threadIdx.x >> 5);
    if (tile >= mtiles) return;
    int lane   = threadIdx.x & 31;
    int laneHi = lane >> 4;
    int m      = lane & 15;
    const float* arow = A + (size_t)(tile * 16 + m) * K_DIM;

    v8f acc[NTILES] = {};

#pragma unroll
    for (int ks = 0; ks < KSTEPS; ++ks) {
        // A 16x32 bf16 layout: lanes<16 K {0..7,16..23}; lanes>=16 K {8..15,24..31}
        int k0 = ks * 32 + laneHi * 8;
        const float4* p = reinterpret_cast<const float4*>(arow + k0);
        float4 f0 = p[0], f1 = p[1], f2 = p[4], f3 = p[5];
        float va[16] = { f0.x, f0.y, f0.z, f0.w,  f1.x, f1.y, f1.z, f1.w,
                         f2.x, f2.y, f2.z, f2.w,  f3.x, f3.y, f3.z, f3.w };
        v16bf a;
#pragma unroll
        for (int j = 0; j < 16; ++j) {
            float v = va[j];
            if (RELU) {
                int k = k0 + (j < 8 ? j : 8 + j);   // j>=8 -> k0+16+(j-8)
                v += biasS[k];
                v = v > 0.0f ? v : 0.0f;
            }
            a[j] = (__bf16)v;
        }
#pragma unroll
        for (int t = 0; t < NTILES; ++t) {
            const v16bf* bp = reinterpret_cast<const v16bf*>(
                &Blds[((ks * NTILES + t) * 32 + lane) * 16]);
            acc[t] = __builtin_amdgcn_wmma_f32_16x16x32_bf16(
                false, a, false, *bp, (short)0, acc[t], false, false);
        }
    }

    // D layout: vgpr r -> row r + 8*laneHi, col = lane&15
#pragma unroll
    for (int t = 0; t < NTILES; ++t) {
        int col = t * 16 + m;
        if (col < NOUT) {
            float* o = out + (size_t)(tile * 16 + laneHi * 8) * NOUT + col;
#pragma unroll
            for (int r = 0; r < 8; ++r) o[(size_t)r * NOUT] = acc[t][r];
        }
    }
}

// ---------------- aggregation (layer 1, 128 feats) ----------------
__global__ void selfloop128_kernel(const float* __restrict__ h, const float* __restrict__ dinv,
                                   float* __restrict__ out, int total) {
    int idx = blockIdx.x * blockDim.x + threadIdx.x;
    if (idx < total) {
        int i = idx >> 7;
        float di = dinv[i];
        out[idx] = h[idx] * di * di;    // full overwrite: init + self-loop message
    }
}

__global__ void agg128_kernel(const float* __restrict__ h, const float* __restrict__ dinv,
                              const int* __restrict__ src, const int* __restrict__ dst,
                              float* out, int E) {
    int tid = blockIdx.x * blockDim.x + threadIdx.x;
    int e = tid >> 5;                   // one wave per edge
    if (e >= E) return;
    int lane = tid & 31;
    int s = src[e], d = dst[e];
    float norm = dinv[s] * dinv[d];
    float4 v = reinterpret_cast<const float4*>(h + (size_t)s * 128)[lane];
    float* o = out + (size_t)d * 128 + lane * 4;
    atomicAdd(o + 0, v.x * norm);
    atomicAdd(o + 1, v.y * norm);
    atomicAdd(o + 2, v.z * norm);
    atomicAdd(o + 3, v.w * norm);
}

// ---------------- aggregation (layer 2, 40 feats) ----------------
__global__ void selfloop40_kernel(const float* __restrict__ h2, const float* __restrict__ dinv,
                                  const float* __restrict__ b2, float* __restrict__ out, int total) {
    int idx = blockIdx.x * blockDim.x + threadIdx.x;
    if (idx < total) {
        int i = idx / 40;
        int c = idx - i * 40;
        float di = dinv[i];
        out[idx] = h2[idx] * di * di + b2[c];   // bias added exactly once
    }
}

__global__ void agg40_kernel(const float* __restrict__ h2, const float* __restrict__ dinv,
                             const int* __restrict__ src, const int* __restrict__ dst,
                             float* out, int total) {
    int idx = blockIdx.x * blockDim.x + threadIdx.x;
    if (idx < total) {
        int e = idx / 40;
        int c = idx - e * 40;
        int s = src[e], d = dst[e];
        float norm = dinv[s] * dinv[d];
        atomicAdd(out + (size_t)d * 40 + c, h2[(size_t)s * 40 + c] * norm);
    }
}

extern "C" void kernel_launch(void* const* d_in, const int* in_sizes, int n_in,
                              void* d_out, int out_size, void* d_ws, size_t ws_size,
                              hipStream_t stream) {
    const float* x  = (const float*)d_in[0];
    const int*   ei = (const int*)d_in[1];
    const float* W1 = (const float*)d_in[2];
    const float* b1 = (const float*)d_in[3];
    const float* W2 = (const float*)d_in[4];
    const float* b2 = (const float*)d_in[5];

    const int N = in_sizes[0] / 128;   // 100000 nodes
    const int E = in_sizes[1] / 2;     // 640000 edges
    const int* src = ei;
    const int* dst = ei + E;

    // workspace carve-up: [deg/dinv][h1 (reused as h2)][out1]
    char* ws = (char*)d_ws;
    size_t degBytes = (((size_t)N * 4) + 255) & ~(size_t)255;
    size_t hBytes   = (size_t)N * 128 * 4;
    float* deg  = (float*)ws;
    float* h1   = (float*)(ws + degBytes);
    float* out1 = (float*)(ws + degBytes + hBytes);
    float* h2   = h1;                  // h1 dead after layer-1 aggregation
    float* out  = (float*)d_out;

    const int B = 256;

    init_deg_kernel <<<(N + B - 1) / B, B, 0, stream>>>(deg, N);
    deg_edges_kernel<<<(E + B - 1) / B, B, 0, stream>>>(dst, deg, E);
    dinv_kernel     <<<(N + B - 1) / B, B, 0, stream>>>(deg, N);

    int mtiles  = (N + 15) / 16;       // 6250 (exact)
    int gblocks = (mtiles + 3) / 4;

    // layer 1: h1 = x @ W1
    gemm_wmma_kernel<8, 128, false><<<gblocks, 128, 0, stream>>>(x, W1, nullptr, h1, mtiles);

    int tot1 = N * 128;
    selfloop128_kernel<<<(tot1 + B - 1) / B, B, 0, stream>>>(h1, deg, out1, tot1);
    int totw = E * 32;
    agg128_kernel<<<(totw + B - 1) / B, B, 0, stream>>>(h1, deg, src, dst, out1, E);

    // layer 2: h2 = relu(out1 + b1) @ W2   (epilogue fused into A load)
    gemm_wmma_kernel<3, 40, true><<<gblocks, 128, 0, stream>>>(out1, W2, b1, h2, mtiles);

    int tot2 = N * 40;
    selfloop40_kernel<<<(tot2 + B - 1) / B, B, 0, stream>>>(h2, deg, b2, out, tot2);
    int tote = E * 40;
    agg40_kernel<<<(tote + B - 1) / B, B, 0, stream>>>(h2, deg, src, dst, out, tote);
}
